// QuChater_20512763805781
// MI455X (gfx1250) — compile-verified, matrix operand order checked by reading, not run on hardware
//
#include <hip/hip_runtime.h>
#include <hip/hip_bf16.h>

// ---------------------------------------------------------------------------
// MI455X (gfx1250) implementation.
//  Kernel 1 (x3): TCN conv layer as WMMA GEMM, M=16 time positions, N=32 ch,
//                 K = 3 taps x 32 ch -> 6 x v_wmma_f32_16x16x32_f16 per tile (2 per tap).
//  Kernel 2:      full 256-step recurrent scan, 16 batch elements / block.
//                 x_t tiles double-buffered in LDS and moved by the Tensor
//                 Data Mover (tensor_load_to_lds, s_wait_tensorcnt 1 for
//                 overlap); gate GEMM = 2 WMMA per wave per step; fast
//                 v_rcp_f32 / v_tanh_f32 transcendentals in the pointwise core.
// ---------------------------------------------------------------------------

typedef __attribute__((ext_vector_type(16))) _Float16 v16h;
typedef __attribute__((ext_vector_type(8)))  float    v8f;
typedef __attribute__((ext_vector_type(4)))  unsigned int u32x4;
typedef __attribute__((ext_vector_type(8)))  int      i32x8;
typedef __attribute__((ext_vector_type(4)))  int      i32x4;

#define B_SZ 4096
#define T_SZ 256
#define H_SZ 32

__device__ __forceinline__ v8f wmma_f16(v16h a, v16h b, v8f c) {
  // (neg_a, A, neg_b, B, c_mod, C, reuse_a, reuse_b)
  return __builtin_amdgcn_wmma_f32_16x16x32_f16(false, a, false, b, (short)0, c,
                                                false, false);
}

// ISA 7.12.2: 16-bit A(16x32)/B(32x16) fragment K index for v16h element e.
__device__ __forceinline__ int kmap(int lane, int e) {
  return ((e >> 3) << 4) + ((lane >> 4) << 3) + (e & 7);
}

// sigmoid via v_exp_f32 + v_rcp_f32 (avoids IEEE division expansion)
__device__ __forceinline__ float sigmf(float x) {
  return __builtin_amdgcn_rcpf(1.0f + __expf(-x));
}

// tanh via CDNA5 v_tanh_f32 when available
__device__ __forceinline__ float fast_tanhf(float x) {
#if __has_builtin(__builtin_amdgcn_tanhf)
  return __builtin_amdgcn_tanhf(x);
#elif __has_builtin(__builtin_amdgcn_tanh_f32)
  return __builtin_amdgcn_tanh_f32(x);
#else
  const float e = __expf(2.0f * x);
  return 1.0f - 2.0f * __builtin_amdgcn_rcpf(e + 1.0f);
#endif
}

// ---------------------------------------------------------------------------
// Tensor Data Mover: 2D f16 tile (rows x width) from global -> LDS.
// D# per ISA 8.3/8.4: group0 = {count=1, lds_addr, global_addr, type=2},
// group1 = {data_size=2B, tensor_dim0=width, tensor_dim1=rows,
//           tile_dim0=width, tile_dim1=rows, dim0_stride=row_stride}.
// Groups 2/3 zero (2D tensor).
// ---------------------------------------------------------------------------
__device__ __forceinline__ void tdm_load_tile_f16(unsigned int lds_off,
                                                  const _Float16* gptr,
                                                  unsigned int width,
                                                  unsigned int rows,
                                                  unsigned int row_stride) {
  unsigned long long ga = (unsigned long long)(size_t)gptr;
  u32x4 g0;
  g0[0] = 1u;                                       // count=1, user mode
  g0[1] = lds_off;                                  // lds_addr (bytes)
  g0[2] = (unsigned int)ga;                         // global_addr[31:0]
  g0[3] = (unsigned int)((ga >> 32) & 0x1FFFFFFull) // global_addr[56:32]
          | (2u << 30);                             // type=2 ("image")
  i32x8 g1;
  g1[0] = (int)(1u << 16);        // workgroup_mask=0, data_size=1 (2 bytes)
  g1[1] = (int)(width << 16);     // tensor_dim0[15:0]
  g1[2] = (int)(rows << 16);      // tensor_dim0[31:16]=0 | tensor_dim1[15:0]
  g1[3] = (int)(width << 16);     // tensor_dim1[31:16]=0 | tile_dim0
  g1[4] = (int)rows;              // tile_dim1 | tile_dim2=0
  g1[5] = (int)row_stride;        // tensor_dim0_stride[31:0] (elements)
  g1[6] = 0;                      // stride hi / dim1_stride lo
  g1[7] = 0;
  i32x4 z4 = {0, 0, 0, 0};
#if __clang_major__ >= 23
  i32x8 z8 = {0, 0, 0, 0, 0, 0, 0, 0};
  __builtin_amdgcn_tensor_load_to_lds(g0, g1, z4, z4, z8, 0);
#else
  __builtin_amdgcn_tensor_load_to_lds(g0, g1, z4, z4, 0);
#endif
}

// ---------------------------------------------------------------------------
// TCN conv layer: out[b,t,n] = relu(sum_{tap,c} W[n,c,tap]*in[b,t+tap-1,c]+bias[n])
// One wave per 16(time) x 32(chan) tile. 8 waves / block. grid = B*T/128.
// ---------------------------------------------------------------------------
template <typename TIN>
__global__ void tcn_conv_kernel(const TIN* __restrict__ in,
                                const float* __restrict__ W,    // [32][32][3]
                                const float* __restrict__ bias, // [32]
                                _Float16* __restrict__ out) {
  const int lane = threadIdx.x & 31;
  const int wave = threadIdx.x >> 5;
  const int tile = blockIdx.x * 8 + wave;   // 16-position tile
  const int pos0 = tile * 16;
  const int bidx = pos0 >> 8;               // /T_SZ (T=256)
  const int t0   = pos0 & (T_SZ - 1);
  const int mrow = lane & 15;               // A-fragment row for this lane

  const TIN* inb = in + (size_t)bidx * T_SZ * H_SZ;

  v8f acc0 = {};
  v8f acc1 = {};

  #pragma unroll
  for (int tap = 0; tap < 3; ++tap) {
    const int tt = t0 + mrow + tap - 1;
    const bool ok = (tt >= 0) && (tt < T_SZ);

    v16h afrag;
    #pragma unroll
    for (int e = 0; e < 16; ++e) {
      const int c = kmap(lane, e);
      float v = ok ? (float)inb[tt * H_SZ + c] : 0.0f;
      afrag[e] = (_Float16)v;
    }
    const int nlo = lane & 15;
    v16h b0, b1;
    #pragma unroll
    for (int e = 0; e < 16; ++e) {
      const int c = kmap(lane, e);
      b0[e] = (_Float16)W[(nlo)      * 96 + c * 3 + tap];
      b1[e] = (_Float16)W[(nlo + 16) * 96 + c * 3 + tap];
    }
    acc0 = wmma_f16(afrag, b0, acc0);
    acc1 = wmma_f16(afrag, b1, acc1);
  }

  const int nlo = lane & 15;
  const float bi0 = bias[nlo];
  const float bi1 = bias[nlo + 16];
  _Float16* outb = out + (size_t)bidx * T_SZ * H_SZ;
  #pragma unroll
  for (int r = 0; r < 8; ++r) {
    const int row = r + ((lane >> 4) << 3);   // C/D layout row
    const int tt  = t0 + row;
    float v0 = acc0[r] + bi0; v0 = v0 > 0.0f ? v0 : 0.0f;
    float v1 = acc1[r] + bi1; v1 = v1 > 0.0f ? v1 : 0.0f;
    outb[tt * H_SZ + nlo]      = (_Float16)v0;
    outb[tt * H_SZ + nlo + 16] = (_Float16)v1;
  }
}

// ---------------------------------------------------------------------------
// Recurrent scan kernel: one block = 16 batch elements, 256 threads (8 waves).
// ---------------------------------------------------------------------------
__global__ void __launch_bounds__(256)
recurrent_kernel(const _Float16* __restrict__ tcn,   // [B][T][32] f16
                 const float* __restrict__ W_ih,     // [128][32]
                 const float* __restrict__ W_hh,     // [128][32]
                 const float* __restrict__ b_ih,     // [128]
                 const float* __restrict__ b_hh,     // [128]
                 const float* __restrict__ q_weights,// [3][6]
                 const float* __restrict__ Wq,       // [32][6]
                 const float* __restrict__ bq,       // [32]
                 const float* __restrict__ Wo,       // [1][32]
                 const float* __restrict__ bo,       // [1]
                 float* __restrict__ outp) {         // [B]
  __shared__ _Float16 sWih[128 * 32];
  __shared__ _Float16 sWhh[128 * 32];
  __shared__ float    sBias[128];
  __shared__ float    sWq[32 * 6];
  __shared__ float    sBq[32];
  __shared__ float    sWo[32];
  __shared__ float    sQc[18];
  __shared__ float    sQs[18];
  __shared__ __align__(16) _Float16 sX[2][16 * 32];   // TDM double buffer
  __shared__ float    sHx[16 * 32];
  __shared__ float    sCx[16 * 32];
  __shared__ float    sHt[16 * 32];
  __shared__ float    sGate[16 * 128];
  __shared__ float    sAr[16 * 64];
  __shared__ float    sAi[16 * 64];
  __shared__ float    sQ[16 * 6];

  const int tid  = threadIdx.x;
  const int lane = tid & 31;
  const int wave = tid >> 5;
  const int bg0  = blockIdx.x * 16;   // first batch element of this block
  const _Float16* xbase = tcn + (size_t)bg0 * (T_SZ * H_SZ);

  // ---- stage weights to LDS (once) ----
  for (int i = tid; i < 128 * 32; i += 256) {
    sWih[i] = (_Float16)W_ih[i];
    sWhh[i] = (_Float16)W_hh[i];
  }
  for (int i = tid; i < 128; i += 256) sBias[i] = b_ih[i] + b_hh[i];
  if (tid < 192) sWq[tid] = Wq[tid];
  if (tid < 32) { sBq[tid] = bq[tid]; sWo[tid] = Wo[tid]; }
  if (tid < 18) {
    const float th = 0.5f * q_weights[tid];
    sQc[tid] = __cosf(th);
    sQs[tid] = __sinf(th);
  }
  for (int i = tid; i < 16 * 32; i += 256) { sHx[i] = 0.0f; sCx[i] = 0.0f; }
  __syncthreads();

  // ---- hoist gate-GEMM B fragments (static weights) out of the t loop ----
  const int n0  = wave * 16;          // gate-column tile for this wave
  const int nlo = lane & 15;
  v16h bI, bH;
  #pragma unroll
  for (int e = 0; e < 16; ++e) {
    const int c = kmap(lane, e);
    bI[e] = sWih[(n0 + nlo) * 32 + c];
    bH[e] = sWhh[(n0 + nlo) * 32 + c];
  }

  const unsigned int sX_lds = (unsigned int)(size_t)&sX[0][0];

  // Prologue: kick off the t=0 tile on the Tensor Data Mover.
  if (wave == 0) {
    tdm_load_tile_f16(sX_lds, xbase, H_SZ, 16, T_SZ * H_SZ);
  }

  // one-qubit rotation: type 0=RX,1=RY,2=RZ.
  auto apply_rot = [&](int wire, int type, int inpIdx, int widx) {
    const int bitpos  = 5 - wire;
    const int lowmask = (1 << bitpos) - 1;
    for (int p = tid; p < 512; p += 256) {
      const int m  = p >> 5;
      const int pl = p & 31;
      const int i0 = ((pl & ~lowmask) << 1) | (pl & lowmask);
      const int i1 = i0 | (1 << bitpos);
      float c, s;
      if (widx >= 0) { c = sQc[widx]; s = sQs[widx]; }
      else {
        const float th = 0.5f * sHt[m * 32 + inpIdx];
        c = __cosf(th); s = __sinf(th);
      }
      const float ar = sAr[m * 64 + i0], ai = sAi[m * 64 + i0];
      const float br = sAr[m * 64 + i1], bi = sAi[m * 64 + i1];
      float n0r, n0i, n1r, n1i;
      if (type == 0) {        // RX: [[c,-is],[-is,c]]
        n0r = c * ar + s * bi;  n0i = c * ai - s * br;
        n1r = s * ai + c * br;  n1i = -s * ar + c * bi;
      } else if (type == 1) { // RY: [[c,-s],[s,c]]
        n0r = c * ar - s * br;  n0i = c * ai - s * bi;
        n1r = s * ar + c * br;  n1i = s * ai + c * bi;
      } else {                // RZ: diag(e^{-i th}, e^{+i th})
        n0r = c * ar + s * ai;  n0i = c * ai - s * ar;
        n1r = c * br - s * bi;  n1i = c * bi + s * br;
      }
      sAr[m * 64 + i0] = n0r; sAi[m * 64 + i0] = n0i;
      sAr[m * 64 + i1] = n1r; sAi[m * 64 + i1] = n1i;
    }
    __syncthreads();
  };

  auto apply_cnot = [&](int cw, int tw) {
    const int cb = 5 - cw, tb = 5 - tw;
    const int lowmask = (1 << tb) - 1;
    for (int p = tid; p < 512; p += 256) {
      const int m  = p >> 5;
      const int pl = p & 31;
      const int i0 = ((pl & ~lowmask) << 1) | (pl & lowmask);
      if (i0 & (1 << cb)) {
        const int i1 = i0 | (1 << tb);
        const float tr = sAr[m * 64 + i0], ti = sAi[m * 64 + i0];
        sAr[m * 64 + i0] = sAr[m * 64 + i1];
        sAi[m * 64 + i0] = sAi[m * 64 + i1];
        sAr[m * 64 + i1] = tr;
        sAi[m * 64 + i1] = ti;
      }
    }
    __syncthreads();
  };

  // ---- sequential scan over T ----
  for (int t = 0; t < T_SZ; ++t) {
    // Issue next tile (overlap with this step's compute), then wait for the
    // current one: completions are in-order, so TENSORcnt<=1 means tile t
    // has landed in LDS.
    if (wave == 0) {
      if (t + 1 < T_SZ) {
        tdm_load_tile_f16(sX_lds + ((unsigned)((t + 1) & 1)) * (16 * 32 * 2),
                          xbase + (t + 1) * H_SZ, H_SZ, 16, T_SZ * H_SZ);
        __builtin_amdgcn_s_wait_tensorcnt(1);
      } else {
        __builtin_amdgcn_s_wait_tensorcnt(0);
      }
    }
    // Warm L2 two steps ahead (one prefetch per row).
    if (wave == 1 && lane < 16 && (t + 2) < T_SZ) {
      __builtin_prefetch(xbase + (size_t)lane * (T_SZ * H_SZ) +
                             (t + 2) * H_SZ, 0, 1);
    }
    __syncthreads();

    const _Float16* xcur = sX[t & 1];

    // gate GEMM: A fragments (X and H), 2 WMMA per wave
    v16h aX, aH;
    #pragma unroll
    for (int e = 0; e < 16; ++e) {
      const int c = kmap(lane, e);
      const int m = lane & 15;
      aX[e] = xcur[m * 32 + c];
      aH[e] = (_Float16)sHx[m * 32 + c];
    }
    v8f acc = {};
    acc = wmma_f16(aX, bI, acc);
    acc = wmma_f16(aH, bH, acc);
    #pragma unroll
    for (int r = 0; r < 8; ++r) {
      const int m = r + ((lane >> 4) << 3);
      sGate[m * 128 + n0 + nlo] = acc[r];
    }
    __syncthreads();

    // LSTM pointwise + logistic map
    for (int i = tid; i < 512; i += 256) {
      const int m = i >> 5, j = i & 31;
      const float ig = sGate[m * 128 + j]       + sBias[j];
      const float fg = sGate[m * 128 + 32 + j]  + sBias[32 + j];
      const float gg = sGate[m * 128 + 64 + j]  + sBias[64 + j];
      const float og = sGate[m * 128 + 96 + j]  + sBias[96 + j];
      const float cnew = sigmf(fg) * sCx[i] + sigmf(ig) * fast_tanhf(gg);
      float hh = sigmf(og) * fast_tanhf(cnew);
      hh = 3.99f * hh * (1.0f - hh);
      sCx[i] = cnew;
      sHt[i] = hh;
    }
    __syncthreads();

    // Henon map on dims 0,1
    if (tid < 16) {
      const float h0 = sHt[tid * 32 + 0];
      const float h1 = sHt[tid * 32 + 1];
      sHt[tid * 32 + 0] = 1.0f - 1.4f * h0 * h0 + h1;
      sHt[tid * 32 + 1] = 0.3f * h0;
    }
    __syncthreads();

    // ---- 6-qubit statevector sim (16 circuits, amps in LDS) ----
    for (int i = tid; i < 1024; i += 256) {
      sAr[i] = ((i & 63) == 0) ? 1.0f : 0.0f;
      sAi[i] = 0.0f;
    }
    __syncthreads();

    #pragma unroll 1
    for (int q = 0; q < 6; ++q) {
      apply_rot(q, 0, q, -1);
      apply_rot(q, 1, (q + 1) % 6, -1);
      apply_rot(q, 2, (q + 2) % 6, -1);
    }
    #pragma unroll 1
    for (int l = 0; l < 3; ++l) {
      for (int q = 0; q < 6; ++q) apply_rot(q, 1, 0, l * 6 + q);
      for (int q = 0; q < 6; ++q) apply_cnot(q, (q + 1) % 6);
    }

    // Z expectations
    if (tid < 96) {
      const int m = tid / 6, qq = tid % 6;
      const int bit = 1 << (5 - qq);
      float accq = 0.0f;
      for (int s = 0; s < 64; ++s) {
        const float pr = sAr[m * 64 + s] * sAr[m * 64 + s] +
                         sAi[m * 64 + s] * sAi[m * 64 + s];
        accq += (s & bit) ? -pr : pr;
      }
      sQ[m * 6 + qq] = accq;
    }
    __syncthreads();

    // hh += q @ Wq^T + bq  -> new hx
    for (int i = tid; i < 512; i += 256) {
      const int m = i >> 5, j = i & 31;
      float a = sHt[i] + sBq[j];
      #pragma unroll
      for (int k = 0; k < 6; ++k) a += sQ[m * 6 + k] * sWq[j * 6 + k];
      sHx[i] = a;
    }
    __syncthreads();
  }

  // output head
  if (tid < 16) {
    float a = bo[0];
    #pragma unroll
    for (int j = 0; j < 32; ++j) a += sHx[tid * 32 + j] * sWo[j];
    outp[bg0 + tid] = sigmf(a);
  }
}

// ---------------------------------------------------------------------------
extern "C" void kernel_launch(void* const* d_in, const int* in_sizes, int n_in,
                              void* d_out, int out_size, void* d_ws, size_t ws_size,
                              hipStream_t stream) {
  (void)in_sizes; (void)n_in; (void)out_size; (void)ws_size;
  const float* x        = (const float*)d_in[0];
  const float* cw1      = (const float*)d_in[1];
  const float* cb1      = (const float*)d_in[2];
  const float* cw2      = (const float*)d_in[3];
  const float* cb2      = (const float*)d_in[4];
  const float* cw3      = (const float*)d_in[5];
  const float* cb3      = (const float*)d_in[6];
  const float* W_ih     = (const float*)d_in[7];
  const float* W_hh     = (const float*)d_in[8];
  const float* b_ih     = (const float*)d_in[9];
  const float* b_hh     = (const float*)d_in[10];
  const float* q_w      = (const float*)d_in[11];
  const float* Wq       = (const float*)d_in[12];
  const float* bq       = (const float*)d_in[13];
  const float* Wo       = (const float*)d_in[14];
  const float* bo       = (const float*)d_in[15];

  _Float16* buf0 = (_Float16*)d_ws;
  _Float16* buf1 = buf0 + (size_t)B_SZ * T_SZ * H_SZ;

  const int nblocks = (B_SZ * T_SZ) / (16 * 8);  // 16 rows/wave, 8 waves/block
  tcn_conv_kernel<float><<<nblocks, 256, 0, stream>>>(x, cw1, cb1, buf0);
  tcn_conv_kernel<_Float16><<<nblocks, 256, 0, stream>>>(buf0, cw2, cb2, buf1);
  tcn_conv_kernel<_Float16><<<nblocks, 256, 0, stream>>>(buf1, cw3, cb3, buf0);

  recurrent_kernel<<<B_SZ / 16, 256, 0, stream>>>(
      buf0, W_ih, W_hh, b_ih, b_hh, q_w, Wq, bq, Wo, bo, (float*)d_out);
}
